// STMVHGNN_22136261443794
// MI455X (gfx1250) — matrix operand order repaired and checked by reading, not compiled, not run on hardware
//
#include <hip/hip_runtime.h>
#include <math.h>

typedef __attribute__((ext_vector_type(16))) _Float16 v16h;
typedef __attribute__((ext_vector_type(8)))  _Float16 v8h;
typedef __attribute__((ext_vector_type(8)))  float    v8f;

#define ACT_NONE 0
#define ACT_TANH 1
#define ACT_RELU 2
#define ACT_SIGM 3

__device__ __forceinline__ void atomAddF(float* p, float v) {
    (void)__hip_atomic_fetch_add(p, v, __ATOMIC_RELAXED, __HIP_MEMORY_SCOPE_AGENT);
}

__device__ __forceinline__ float fast_sigmoid(float v) {
    return __builtin_amdgcn_rcpf(1.0f + __expf(-v));        // v_rcp_f32 + v_exp_f32
}
__device__ __forceinline__ float fast_tanh(float v) {
    // tanh(x) = 1 - 2/(exp(2x)+1)
    return 1.0f - 2.0f * __builtin_amdgcn_rcpf(__expf(2.0f * v) + 1.0f);
}

// ---------------------------------------------------------------------------
// WMMA GEMM, compile-time shapes:  C[M x NT] = act( A[M x KT] @ B[KT x NT] + bias )
// CONCAT: A = [A0 (width kSplit) | A1 (width KT-kSplit)].
// Block = one 16-row stripe; wave w -> 16x16 tile at column w*16.
// A tile (16xKT) and B (KT x NT, transposed) staged in LDS as f16 with +8-half
// row padding (kills stride-256B bank conflicts). Fragment builds are pairs of
// 16B ds_load_b128 matching the CDNA5 A-16x32 / B-32x16 half-wave K layouts.
// ---------------------------------------------------------------------------
template <int KT, int NT, bool CONCAT>
__global__ __launch_bounds__((NT / 16) * 32) void gemm_wmma_f16(
    const float* __restrict__ A0, const float* __restrict__ A1, int kSplit,
    const float* __restrict__ B, const float* __restrict__ bias,
    float* __restrict__ C, int act)
{
    constexpr int THREADS = (NT / 16) * 32;
    constexpr int LDK = KT + 8;                  // padded row stride (halfs)
    __shared__ _Float16 lds_a[16 * LDK];
    __shared__ _Float16 lds_b[NT * LDK];

    const int tid   = threadIdx.x;
    const int lane  = tid & 31;
    const int wave  = tid >> 5;
    const int tileM = blockIdx.x;

    // ---- stage A tile (16 x KT) row-major f16 ----
    for (int idx = tid; idx < 16 * KT; idx += THREADS) {
        const int r = idx / KT, k = idx % KT;    // KT power of two -> shifts
        float v;
        if (CONCAT) {
            v = (k < kSplit)
                    ? A0[(size_t)(tileM * 16 + r) * kSplit + k]
                    : A1[(size_t)(tileM * 16 + r) * (KT - kSplit) + (k - kSplit)];
        } else {
            v = A0[(size_t)(tileM * 16 + r) * KT + k];
        }
        lds_a[r * LDK + k] = (_Float16)v;
    }
    // ---- stage B (KT x NT) transposed -> lds_b[n*LDK + k] ----
    for (int idx = tid; idx < KT * NT; idx += THREADS) {
        const int k = idx / NT, n = idx % NT;
        lds_b[n * LDK + k] = (_Float16)B[idx];
    }
    __syncthreads();

    const int half = lane >> 4;                  // 0: lanes 0-15, 1: lanes 16-31
    const int l15  = lane & 15;
    const int half8  = half * 8;
    const int half16 = half * 16;
    const int col = wave * 16 + l15;             // B/C column owned by this lane

    v8f acc = {};
#pragma unroll
    for (int k0 = 0; k0 < KT; k0 += 32) {
        // A 16x32 fragment: lanes0-15 K{0..7,16..23}, lanes16-31 K{8..15,24..31}
        const v8h alo = *(const v8h*)(lds_a + l15 * LDK + k0 + half8);
        const v8h ahi = *(const v8h*)(lds_a + l15 * LDK + k0 + 16 + half8);
        const v16h av = __builtin_shufflevector(alo, ahi,
            0, 1, 2, 3, 4, 5, 6, 7, 8, 9, 10, 11, 12, 13, 14, 15);
        // B 32x16 fragment: lanes0-15 K{0..7,16..23}, lanes16-31 K{8..15,24..31}
        const v8h blo = *(const v8h*)(lds_b + col * LDK + k0 + half8);
        const v8h bhi = *(const v8h*)(lds_b + col * LDK + k0 + 16 + half8);
        const v16h bv = __builtin_shufflevector(blo, bhi,
            0, 1, 2, 3, 4, 5, 6, 7, 8, 9, 10, 11, 12, 13, 14, 15);
        acc = __builtin_amdgcn_wmma_f32_16x16x32_f16(
                  false, av, false, bv, (short)0, acc, false, false);
    }

#pragma unroll
    for (int i = 0; i < 8; ++i) {
        const int r = tileM * 16 + i + half8;    // C: VGPR i -> row i (+8 upper half)
        float v = acc[i];
        if (bias) v += bias[col];
        if      (act == ACT_TANH) v = fast_tanh(v);
        else if (act == ACT_RELU) v = fmaxf(v, 0.0f);
        else if (act == ACT_SIGM) v = fast_sigmoid(v);
        C[(size_t)r * NT + col] = v;
    }
}

// ---------------------------------------------------------------------------
// Zero fill
// ---------------------------------------------------------------------------
__global__ void zero_kernel(float* __restrict__ p, long long n) {
    long long i = (long long)blockIdx.x * blockDim.x + threadIdx.x;
    long long s = (long long)gridDim.x * blockDim.x;
    for (; i < n; i += s) p[i] = 0.0f;
}

// ---------------------------------------------------------------------------
// Degrees: D[node] += w[edge], Bcnt[edge] += 1 per incidence
// ---------------------------------------------------------------------------
__global__ void degree_kernel(const int* __restrict__ nodes, const int* __restrict__ edges,
                              const float* __restrict__ we,
                              float* __restrict__ D, float* __restrict__ Bcnt, int M)
{
    int i = blockIdx.x * blockDim.x + threadIdx.x;
    int s = gridDim.x * blockDim.x;
    for (; i < M; i += s) {
        atomAddF(&D[nodes[i]], we[edges[i]]);
        atomAddF(&Bcnt[edges[i]], 1.0f);
    }
}

// ---------------------------------------------------------------------------
// Scatter node features into hyperedges: ef[e] += xt[n]  (wave per incidence)
// ---------------------------------------------------------------------------
__global__ void scatter_n2e(const int* __restrict__ nodes, const int* __restrict__ edges,
                            const float* __restrict__ xt, float* __restrict__ ef, int M)
{
    const int lane = threadIdx.x & 31;
    int w = (blockIdx.x * blockDim.x + threadIdx.x) >> 5;
    int ws = (gridDim.x * blockDim.x) >> 5;
    for (; w < M; w += ws) {
        const int n = nodes[w], e = edges[w];
        const float4 v = *(const float4*)(xt + (size_t)n * 128 + lane * 4);
        float* dst = ef + (size_t)e * 128 + lane * 4;
        atomAddF(dst + 0, v.x); atomAddF(dst + 1, v.y);
        atomAddF(dst + 2, v.z); atomAddF(dst + 3, v.w);
    }
}

// ---------------------------------------------------------------------------
// Scatter hyperedge features back to nodes: z[n] += w_e * ef[e]
// ---------------------------------------------------------------------------
__global__ void scatter_e2n(const int* __restrict__ nodes, const int* __restrict__ edges,
                            const float* __restrict__ we, const float* __restrict__ ef,
                            float* __restrict__ z, int M)
{
    const int lane = threadIdx.x & 31;
    int w = (blockIdx.x * blockDim.x + threadIdx.x) >> 5;
    int ws = (gridDim.x * blockDim.x) >> 5;
    for (; w < M; w += ws) {
        const int n = nodes[w], e = edges[w];
        const float wv = we[e];
        const float4 v = *(const float4*)(ef + (size_t)e * 128 + lane * 4);
        float* dst = z + (size_t)n * 128 + lane * 4;
        atomAddF(dst + 0, wv * v.x); atomAddF(dst + 1, wv * v.y);
        atomAddF(dst + 2, wv * v.z); atomAddF(dst + 3, wv * v.w);
    }
}

// ef[r][c] *= (Bcnt[r] > 0 ? 1/Bcnt[r] : 0)
__global__ void scale_rows(float* __restrict__ ef, const float* __restrict__ Bcnt, long long n)
{
    long long i = (long long)blockIdx.x * blockDim.x + threadIdx.x;
    long long s = (long long)gridDim.x * blockDim.x;
    for (; i < n; i += s) {
        const float c = Bcnt[i >> 7];
        ef[i] *= (c > 0.0f) ? __builtin_amdgcn_rcpf(c) : 0.0f;
    }
}

// z = leaky_relu(Dinv * z + b[c], 0.2)
__global__ void finalize_z(float* __restrict__ z, const float* __restrict__ D,
                           const float* __restrict__ b, long long n)
{
    long long i = (long long)blockIdx.x * blockDim.x + threadIdx.x;
    long long s = (long long)gridDim.x * blockDim.x;
    for (; i < n; i += s) {
        const float d = D[i >> 7];
        const float dinv = (d > 0.0f) ? __builtin_amdgcn_rcpf(d) : 0.0f;
        const float v = dinv * z[i] + b[i & 127];
        z[i] = (v > 0.0f) ? v : 0.2f * v;
    }
}

// zs = g*zs + (1-g)*zb
__global__ void fuse_kernel(float* __restrict__ zs, const float* __restrict__ g,
                            const float* __restrict__ zb, long long n)
{
    long long i = (long long)blockIdx.x * blockDim.x + threadIdx.x;
    long long s = (long long)gridDim.x * blockDim.x;
    for (; i < n; i += s) {
        const float gv = g[i];
        zs[i] = gv * zs[i] + (1.0f - gv) * zb[i];
    }
}

// ---------------------------------------------------------------------------
// Row LayerNorm (H=128) + leaky-relu(slope) in place. slope==0 -> ReLU.
// One wave per row, lane handles 4 channels.
// ---------------------------------------------------------------------------
__global__ void ln_act(float* __restrict__ X, const float* __restrict__ g,
                       const float* __restrict__ b, int N, float slope)
{
    const int lane = threadIdx.x & 31;
    int row = (blockIdx.x * blockDim.x + threadIdx.x) >> 5;
    int rs  = (gridDim.x * blockDim.x) >> 5;
    for (; row < N; row += rs) {
        float* p = X + (size_t)row * 128 + lane * 4;
        float4 v = *(float4*)p;
        float s = v.x + v.y + v.z + v.w;
#pragma unroll
        for (int o = 16; o >= 1; o >>= 1) s += __shfl_xor(s, o, 32);
        const float mean = s * (1.0f / 128.0f);
        const float dx = v.x - mean, dy = v.y - mean, dz = v.z - mean, dw = v.w - mean;
        float q = dx * dx + dy * dy + dz * dz + dw * dw;
#pragma unroll
        for (int o = 16; o >= 1; o >>= 1) q += __shfl_xor(q, o, 32);
        const float inv = rsqrtf(q * (1.0f / 128.0f) + 1e-5f);
        const int c = lane * 4;
        float o0 = dx * inv * g[c + 0] + b[c + 0];
        float o1 = dy * inv * g[c + 1] + b[c + 1];
        float o2 = dz * inv * g[c + 2] + b[c + 2];
        float o3 = dw * inv * g[c + 3] + b[c + 3];
        o0 = (o0 > 0.0f) ? o0 : slope * o0;
        o1 = (o1 > 0.0f) ? o1 : slope * o1;
        o2 = (o2 > 0.0f) ? o2 : slope * o2;
        o3 = (o3 > 0.0f) ? o3 : slope * o3;
        *(float4*)p = make_float4(o0, o1, o2, o3);
    }
}

// out[row*stride] = dot(X[row,:128], w) + bias[0]   (wave per row)
__global__ void rowdot(const float* __restrict__ X, const float* __restrict__ w,
                       const float* __restrict__ bias, float* __restrict__ out,
                       int stride, int N)
{
    const int lane = threadIdx.x & 31;
    int row = (blockIdx.x * blockDim.x + threadIdx.x) >> 5;
    int rs  = (gridDim.x * blockDim.x) >> 5;
    for (; row < N; row += rs) {
        const float4 v  = *(const float4*)(X + (size_t)row * 128 + lane * 4);
        const float4 ww = *(const float4*)(w + lane * 4);
        float s = v.x * ww.x + v.y * ww.y + v.z * ww.z + v.w * ww.w;
#pragma unroll
        for (int o = 16; o >= 1; o >>= 1) s += __shfl_xor(s, o, 32);
        if (lane == 0) out[(size_t)row * stride] = s + bias[0];
    }
}

// softmax over the 3 per-view scores, then zs = sum_v attn_v * z_v  (wave per row)
__global__ void attn_combine(const float* __restrict__ scores,
                             const float* __restrict__ z0, const float* __restrict__ z1,
                             const float* __restrict__ z2, float* __restrict__ zs, int N)
{
    const int lane = threadIdx.x & 31;
    int row = (blockIdx.x * blockDim.x + threadIdx.x) >> 5;
    int rs  = (gridDim.x * blockDim.x) >> 5;
    for (; row < N; row += rs) {
        const float s0 = scores[row * 3 + 0];
        const float s1 = scores[row * 3 + 1];
        const float s2 = scores[row * 3 + 2];
        const float m  = fmaxf(s0, fmaxf(s1, s2));
        const float e0 = __expf(s0 - m), e1 = __expf(s1 - m), e2 = __expf(s2 - m);
        const float inv = __builtin_amdgcn_rcpf(e0 + e1 + e2);
        const float a0 = e0 * inv, a1 = e1 * inv, a2 = e2 * inv;
        const size_t off = (size_t)row * 128 + lane * 4;
        const float4 v0 = *(const float4*)(z0 + off);
        const float4 v1 = *(const float4*)(z1 + off);
        const float4 v2 = *(const float4*)(z2 + off);
        float4 o;
        o.x = a0 * v0.x + a1 * v1.x + a2 * v2.x;
        o.y = a0 * v0.y + a1 * v1.y + a2 * v2.y;
        o.z = a0 * v0.z + a1 * v1.z + a2 * v2.z;
        o.w = a0 * v0.w + a1 * v1.w + a2 * v2.w;
        *(float4*)(zs + off) = o;
    }
}

// ---------------------------------------------------------------------------
// Host orchestration
// ---------------------------------------------------------------------------
extern "C" void kernel_launch(void* const* d_in, const int* in_sizes, int n_in,
                              void* d_out, int out_size, void* d_ws, size_t ws_size,
                              hipStream_t stream)
{
    const float* x      = (const float*)d_in[0];
    const int*   e_v[3] = { (const int*)d_in[1], (const int*)d_in[2], (const int*)d_in[3] };
    const float* w_v[3] = { (const float*)d_in[4], (const float*)d_in[5], (const float*)d_in[6] };
    const float* W_v[3] = { (const float*)d_in[7], (const float*)d_in[9], (const float*)d_in[11] };
    const float* b_v[3] = { (const float*)d_in[8], (const float*)d_in[10], (const float*)d_in[12] };
    const float* Wa1 = (const float*)d_in[13]; const float* ba1 = (const float*)d_in[14];
    const float* Wa2 = (const float*)d_in[15]; const float* ba2 = (const float*)d_in[16];
    const float* Wb  = (const float*)d_in[17]; const float* bb  = (const float*)d_in[18];
    const float* lnbg = (const float*)d_in[19]; const float* lnbb = (const float*)d_in[20];
    const float* Wg1 = (const float*)d_in[21]; const float* bg1 = (const float*)d_in[22];
    const float* Wg2 = (const float*)d_in[23]; const float* bg2 = (const float*)d_in[24];
    const float* Wc1 = (const float*)d_in[25]; const float* bc1 = (const float*)d_in[26];
    const float* lncg = (const float*)d_in[27]; const float* lncb = (const float*)d_in[28];
    const float* Wc2 = (const float*)d_in[29]; const float* bc2 = (const float*)d_in[30];

    const int N = in_sizes[0] / 128;     // 100000 (multiple of 16)
    const int M = in_sizes[1] / 2;       // 800000
    const int E = in_sizes[4];           // 100000

    // workspace layout
    char* ws = (char*)d_ws;
    size_t off = 0;
    float* xt = (float*)(ws + off); off += (size_t)N * 128 * 4;   // reused: xt / tanh-tile / g
    float* ef = (float*)(ws + off); off += (size_t)E * 128 * 4;   // reused: edge feats / classifier hidden
    float* z0 = (float*)(ws + off); off += (size_t)N * 128 * 4;
    float* z1 = (float*)(ws + off); off += (size_t)N * 128 * 4;
    float* z2 = (float*)(ws + off); off += (size_t)N * 128 * 4;
    float* zviews[3] = { z0, z1, z2 };
    float* zs = (float*)(ws + off); off += (size_t)N * 128 * 4;   // z_spatial -> z_fused
    float* zb = (float*)(ws + off); off += (size_t)N * 128 * 4;   // z_behavioral
    float* se = (float*)(ws + off); off += (size_t)N * 64 * 4;    // gating hidden
    float* scores = (float*)(ws + off); off += (size_t)N * 3 * 4;
    float* Dd = (float*)(ws + off); off += (size_t)N * 4;
    float* Bc = (float*)(ws + off); off += (size_t)E * 4;

    const long long NH = (long long)N * 128;
    const long long EH = (long long)E * 128;

    const dim3 blk(256), blk128(128);
    const dim3 gGemm((unsigned)(N / 16));
    const dim3 gEW(8192), gWAVE(4096), gSC(16384), gDEG(2048);

    // ---- per-view hypergraph convolutions ----
    for (int v = 0; v < 3; ++v) {
        const int* nodes = e_v[v];
        const int* edges = e_v[v] + M;
        zero_kernel<<<dim3(512), blk, 0, stream>>>(Dd, N);
        zero_kernel<<<dim3(512), blk, 0, stream>>>(Bc, E);
        zero_kernel<<<gEW, blk, 0, stream>>>(ef, EH);
        zero_kernel<<<gEW, blk, 0, stream>>>(zviews[v], NH);
        degree_kernel<<<gDEG, blk, 0, stream>>>(nodes, edges, w_v[v], Dd, Bc, M);
        // xt = x @ W_v
        gemm_wmma_f16<128, 128, false><<<gGemm, blk, 0, stream>>>(
            x, nullptr, 128, W_v[v], nullptr, xt, ACT_NONE);
        scatter_n2e<<<gSC, blk, 0, stream>>>(nodes, edges, xt, ef, M);
        scale_rows<<<gEW, blk, 0, stream>>>(ef, Bc, EH);
        scatter_e2n<<<gSC, blk, 0, stream>>>(nodes, edges, w_v[v], ef, zviews[v], M);
        finalize_z<<<gEW, blk, 0, stream>>>(zviews[v], Dd, b_v[v], NH);
    }

    // ---- view attention ----
    for (int v = 0; v < 3; ++v) {
        gemm_wmma_f16<128, 128, false><<<gGemm, blk, 0, stream>>>(
            zviews[v], nullptr, 128, Wa1, ba1, xt, ACT_TANH);
        rowdot<<<gWAVE, blk, 0, stream>>>(xt, Wa2, ba2, scores + v, 3, N);
    }
    attn_combine<<<gWAVE, blk, 0, stream>>>(scores, z0, z1, z2, zs, N);

    // ---- behavioral branch: zb = lrelu(LN(x @ Wb + bb)) ----
    gemm_wmma_f16<128, 128, false><<<gGemm, blk, 0, stream>>>(
        x, nullptr, 128, Wb, bb, zb, ACT_NONE);
    ln_act<<<gWAVE, blk, 0, stream>>>(zb, lnbg, lnbb, N, 0.2f);

    // ---- gating: se = relu([zs,zb]@Wg1+bg1); g = sigmoid(se@Wg2+bg2); fuse ----
    gemm_wmma_f16<256, 64, true><<<gGemm, blk128, 0, stream>>>(
        zs, zb, 128, Wg1, bg1, se, ACT_RELU);
    gemm_wmma_f16<64, 128, false><<<gGemm, blk, 0, stream>>>(
        se, nullptr, 64, Wg2, bg2, xt /*g*/, ACT_SIGM);
    fuse_kernel<<<gEW, blk, 0, stream>>>(zs, xt, zb, NH);

    // ---- classifier: h = relu(LN([zs,x]@Wc1+bc1)); out = h@Wc2 + bc2 ----
    gemm_wmma_f16<256, 128, true><<<gGemm, blk, 0, stream>>>(
        zs, x, 128, Wc1, bc1, ef /*h*/, ACT_NONE);
    ln_act<<<gWAVE, blk, 0, stream>>>(ef, lncg, lncb, N, 0.0f);
    rowdot<<<gWAVE, blk, 0, stream>>>(ef, Wc2, bc2, (float*)d_out, 1, N);
}
